// Cross_mamba_both_11811160064647
// MI455X (gfx1250) — matrix operand order repaired and checked by reading, not compiled, not run on hardware
//
#include <hip/hip_runtime.h>
#include <hip/hip_bf16.h>
#include <stdint.h>

// ---------------------------------------------------------------------------
// Model dims (from reference)
// ---------------------------------------------------------------------------
#define BSZ     4
#define SEQ     153          // 1 cls + 8 categ + 16 numeric + 128 img tokens
#define ROWS    (BSZ*SEQ)    // 612
#define DIM     512
#define DINNER  1024
#define DSTATE  16
#define NKEY    192          // 96 mri channels + 96 pet channels
#define DCROSS  25600        // 160*160
#define FFH     2048
#define KVSPLIT 4

// CDNA5 async global->LDS copy path (guarded: falls back to register staging)
#if defined(__has_builtin)
#if __has_builtin(__builtin_amdgcn_global_load_async_to_lds_b128) && \
    __has_builtin(__builtin_amdgcn_s_wait_asynccnt)
#define USE_ASYNC_LDS 1
#endif
#endif
#ifndef USE_ASYNC_LDS
#define USE_ASYNC_LDS 0
#endif

typedef __attribute__((ext_vector_type(16))) __bf16 v16bf;
typedef __attribute__((ext_vector_type(8)))  float  v8f;
typedef __attribute__((ext_vector_type(4)))  int    v4i;

union BFrag { uint32_t u[8]; v16bf v; };

__device__ __forceinline__ unsigned short f2bf(float f) {
  unsigned int u = __float_as_uint(f);
  unsigned int r = u + 0x7FFFu + ((u >> 16) & 1u);   // round-to-nearest-even
  return (unsigned short)(r >> 16);
}

// pack two f32 -> two bf16 (RNE) in one dword: lo in [15:0], hi in [31:16]
__device__ __forceinline__ uint32_t pack_bf16(float lo, float hi) {
  uint32_t ul = __float_as_uint(lo);
  uint32_t uh = __float_as_uint(hi);
  ul += 0x7FFFu + ((ul >> 16) & 1u);
  uh += 0x7FFFu + ((uh >> 16) & 1u);
  // v_perm_b32: bytes 0-3 select from src1(ul), 4-7 from src0(uh)
  return __builtin_amdgcn_perm(uh, ul, 0x07060302u);
}

__device__ __forceinline__ uint2 pack_bf16x4(float4 f) {
  uint2 r;
  r.x = pack_bf16(f.x, f.y);
  r.y = pack_bf16(f.z, f.w);
  return r;
}

// ---------------------------------------------------------------------------
// bf16 WMMA GEMM:  C[M,N] = A_bf16[M,K] @ B_f32[N,K]^T (+bias)(+residual)
// 256 threads = 8 waves; block tile 64(M) x 64(N) x 64(K), double-buffered LDS.
// A tile: async DMA global->LDS (raw bf16, no conversion) when available.
// B tile: global->VGPR->convert->LDS.
// Requires: N % 64 == 0, K % (64*kslices) == 0. M guarded by clamping A row
// loads (padded rows only feed C rows the epilogue never stores).
// kslices > 1: blockIdx.z selects a K slice; raw partials go to Cout + z*M*N.
// ---------------------------------------------------------------------------
__global__ __launch_bounds__(256) void gemm_bf16_wmma(
    const unsigned short* __restrict__ A, int lda,
    const float* __restrict__ Bw, int ldb,
    const float* __restrict__ bias,
    const float* __restrict__ Cres,
    float* __restrict__ Cout,
    int M, int N, int K, int kslices)
{
  __shared__ __align__(16) uint32_t As[2][64][36];   // [buf][m][k-pair], 144B row
  __shared__ __align__(16) uint32_t Bs[2][64][36];   // [buf][n][k-pair]

  const int t    = threadIdx.x;
  const int lane = t & 31, wave = t >> 5;
  const int wm = wave >> 1, wn = wave & 1;
  const int m0 = blockIdx.y * 64, n0 = blockIdx.x * 64;
  const int row = lane & 15, half = lane >> 4;

  const int klen = K / kslices;
  const int kbeg = blockIdx.z * klen;
  const int kend = kbeg + klen;

  // A staging: 4 lanes cover one 128B row slice; 2 x 16B per thread
  const int a_m = t >> 2, a_q = t & 3;
  // B staging: 16 lanes cover one 256B row slice; 4 float4 per thread
  const int b_n = t >> 4, b_q = t & 15;

  int amg = m0 + a_m; if (amg >= M) amg = 0;         // clamp: no divergent loads
  const unsigned short* aRow = A + (size_t)amg * lda + 8 * a_q;
  const float* bRow0 = Bw + (size_t)(n0 + b_n)      * ldb + 4 * b_q;
  const float* bRow1 = Bw + (size_t)(n0 + b_n + 16) * ldb + 4 * b_q;
  const float* bRow2 = Bw + (size_t)(n0 + b_n + 32) * ldb + 4 * b_q;
  const float* bRow3 = Bw + (size_t)(n0 + b_n + 48) * ldb + 4 * b_q;

  // staged tile registers (named scalars: keep out of scratch)
  uint4  ar0, ar1;                                   // fallback path only
  float4 br0, br1, br2, br3;
  (void)ar0; (void)ar1;

#define LOAD_B(K0)                                   \
  do {                                               \
    br0 = *(const float4*)(bRow0 + (K0));            \
    br1 = *(const float4*)(bRow1 + (K0));            \
    br2 = *(const float4*)(bRow2 + (K0));            \
    br3 = *(const float4*)(bRow3 + (K0));            \
  } while (0)

#define STORE_B(BUF)                                      \
  do {                                                    \
    *(uint2*)&Bs[BUF][b_n     ][2 * b_q] = pack_bf16x4(br0); \
    *(uint2*)&Bs[BUF][b_n + 16][2 * b_q] = pack_bf16x4(br1); \
    *(uint2*)&Bs[BUF][b_n + 32][2 * b_q] = pack_bf16x4(br2); \
    *(uint2*)&Bs[BUF][b_n + 48][2 * b_q] = pack_bf16x4(br3); \
  } while (0)

#if USE_ASYNC_LDS
  // Builtin signature (from hipcc diagnostic): param 1 is v4i* in the device
  // (global, AS1) address space; param 2 is the LDS (AS3) v4i*; then two
  // constant ints (offset, cpol). INST_OFFSET is added to BOTH the LDS and
  // global address (ISA §10/§15.18); our LDS row layout mirrors the global
  // layout, so quad 1 is base+64B on both sides.
#define ISSUE_A(K0, BUF)                                                        \
  do {                                                                          \
    __builtin_amdgcn_global_load_async_to_lds_b128(                             \
        (__attribute__((address_space(1))) v4i*)(aRow + (K0)),                  \
        (__attribute__((address_space(3))) v4i*)&As[BUF][a_m][4 * a_q], 0, 0);  \
    __builtin_amdgcn_global_load_async_to_lds_b128(                             \
        (__attribute__((address_space(1))) v4i*)(aRow + (K0)),                  \
        (__attribute__((address_space(3))) v4i*)&As[BUF][a_m][4 * a_q], 64, 0); \
  } while (0)
#else
#define LOAD_A(K0)                                   \
  do {                                               \
    ar0 = *(const uint4*)(aRow + (K0));              \
    ar1 = *(const uint4*)(aRow + (K0) + 32);         \
  } while (0)
#define STORE_A(BUF)                                 \
  do {                                               \
    *(uint4*)&As[BUF][a_m][4 * a_q]      = ar0;      \
    *(uint4*)&As[BUF][a_m][4 * a_q + 16] = ar1;      \
  } while (0)
#endif

  v8f acc0 = {}; v8f acc1 = {};

  int buf = 0;
#if USE_ASYNC_LDS
  LOAD_B(kbeg);
  ISSUE_A(kbeg, 0);
#else
  LOAD_B(kbeg);
  LOAD_A(kbeg);
#endif
  for (int k0 = kbeg; k0 < kend; k0 += 64) {
#if USE_ASYNC_LDS
    STORE_B(buf);
    __builtin_amdgcn_s_wait_asynccnt(0);             // A tile landed in LDS
#else
    STORE_A(buf);
    STORE_B(buf);
#endif
    __syncthreads();
    if (k0 + 64 < kend) {                            // prefetch next tile
#if USE_ASYNC_LDS
      LOAD_B(k0 + 64);
      ISSUE_A(k0 + 64, buf ^ 1);
#else
      LOAD_B(k0 + 64);
      LOAD_A(k0 + 64);
#endif
    }
    #pragma unroll
    for (int ks = 0; ks < 2; ++ks) {
      BFrag a, b0, b1;
      #pragma unroll
      for (int v = 0; v < 8; ++v) {
        const int kpa = ks * 16 + (half << 3) + v;
        const int kpb = ks * 16 + (half << 3) + v;
        a.u[v]  = As[buf][(wm << 4) + row][kpa];
        b0.u[v] = Bs[buf][(wn << 5) + row][kpb];
        b1.u[v] = Bs[buf][(wn << 5) + 16 + row][kpb];
      }
      acc0 = __builtin_amdgcn_wmma_f32_16x16x32_bf16(false, a.v, false, b0.v,
                                                     (short)0, acc0, false, false);
      acc1 = __builtin_amdgcn_wmma_f32_16x16x32_bf16(false, a.v, false, b1.v,
                                                     (short)0, acc1, false, false);
    }
    buf ^= 1;
  }
#undef LOAD_B
#undef STORE_B
#if USE_ASYNC_LDS
#undef ISSUE_A
#else
#undef LOAD_A
#undef STORE_A
#endif

  // epilogue: C layout lanes 0-15 -> m=r, lanes 16-31 -> m=r+8, n=lane&15
  float* Cdst = (kslices == 1) ? Cout : Cout + (size_t)blockIdx.z * M * N;
  #pragma unroll
  for (int r = 0; r < 8; ++r) {
    const int mg = m0 + (wm << 4) + r + (half << 3);
    if (mg < M) {
      const int ng0 = n0 + (wn << 5) + row;
      const int ng1 = ng0 + 16;
      size_t base = (size_t)mg * N;
      float o0 = acc0[r], o1 = acc1[r];
      if (kslices == 1) {
        if (bias) { o0 += bias[ng0]; o1 += bias[ng1]; }
        if (Cres) { o0 += Cres[base + ng0]; o1 += Cres[base + ng1]; }
      }
      Cdst[base + ng0] = o0;
      Cdst[base + ng1] = o1;
    }
  }
}

// ---------------------------------------------------------------------------
// Deterministic split-K reduction: out = sum_z part[z] (+bias)(+res)
// ---------------------------------------------------------------------------
__global__ void splitk_reduce_kernel(const float* __restrict__ part,
                                     const float* __restrict__ bias,
                                     const float* __restrict__ res,
                                     float* __restrict__ out,
                                     int MN, int N, int S)
{
  int idx = blockIdx.x * blockDim.x + threadIdx.x;
  if (idx >= MN) return;
  float s = 0.f;
  for (int z = 0; z < S; ++z) s += part[(size_t)z * MN + idx];
  if (bias) s += bias[idx % N];
  if (res)  s += res[idx];
  out[idx] = s;
}

// ---------------------------------------------------------------------------
// Token embedding: x[b,l,d]
// ---------------------------------------------------------------------------
__global__ void embed_kernel(const int* __restrict__ x_categ,
                             const float* __restrict__ x_numer,
                             const float* __restrict__ feature_img,
                             const float* __restrict__ categ_embed,
                             const float* __restrict__ num_w,
                             const float* __restrict__ num_b,
                             const float* __restrict__ cls,
                             float* __restrict__ x)
{
  int idx = blockIdx.x * blockDim.x + threadIdx.x;
  if (idx >= ROWS * DIM) return;
  int d = idx & (DIM - 1);
  int bl = idx >> 9;
  int b = bl / SEQ, l = bl % SEQ;
  float v;
  if (l == 0) {
    v = cls[d];
  } else if (l <= 8) {
    int i = l - 1;
    int tok = x_categ[b * 8 + i] + 2 + 10 * i;   // OFFSETS = 2 + 10*i
    v = categ_embed[tok * DIM + d];
  } else if (l <= 24) {
    int j = l - 9;
    v = x_numer[b * 16 + j] * num_w[j * DIM + d] + num_b[j * DIM + d];
  } else {
    v = feature_img[((b * 128) + (l - 25)) * DIM + d];
  }
  x[idx] = v;
}

// ---------------------------------------------------------------------------
// RMSNorm row -> bf16 (GEMM A operand)
// ---------------------------------------------------------------------------
__global__ void rmsnorm_kernel(const float* __restrict__ x,
                               const float* __restrict__ w,
                               unsigned short* __restrict__ out)
{
  __shared__ float red[256];
  __shared__ float scale;
  int bl = blockIdx.x, t = threadIdx.x;
  float s = 0.f;
  for (int i = t; i < DIM; i += 256) { float v = x[bl * DIM + i]; s += v * v; }
  red[t] = s;
  __syncthreads();
  for (int off = 128; off > 0; off >>= 1) {
    if (t < off) red[t] += red[t + off];
    __syncthreads();
  }
  if (t == 0) scale = rsqrtf(red[0] / (float)DIM + 1e-5f);
  __syncthreads();
  float sc = scale;
  for (int i = t; i < DIM; i += 256)
    out[bl * DIM + i] = f2bf(x[bl * DIM + i] * sc * w[i]);
}

// ---------------------------------------------------------------------------
// Depthwise causal conv1d (width 4, left pad 3) + bias + SiLU on xi half of xz
// ---------------------------------------------------------------------------
__global__ void conv_silu_kernel(const float* __restrict__ xz,
                                 const float* __restrict__ conv_w,
                                 const float* __restrict__ conv_b,
                                 float* __restrict__ xi)
{
  int idx = blockIdx.x * blockDim.x + threadIdx.x;
  if (idx >= ROWS * DINNER) return;
  int d = idx & (DINNER - 1);
  int bl = idx >> 10;
  int l = bl % SEQ;
  float acc = conv_b[d];
  #pragma unroll
  for (int j = 0; j < 4; ++j) {
    int li = l + j - 3;
    if (li >= 0) acc += xz[(size_t)(bl + j - 3) * 2048 + d] * conv_w[d * 4 + j];
  }
  xi[idx] = acc / (1.f + __expf(-acc));   // silu
}

// ---------------------------------------------------------------------------
// dbc[bl, 0:64] = xi_row(1024) @ xproj_w[64,1024]^T   (block per row)
// ---------------------------------------------------------------------------
__global__ void xproj_kernel(const float* __restrict__ xi,
                             const float* __restrict__ xproj_w,
                             float* __restrict__ dbc)
{
  __shared__ float srow[DINNER];
  __shared__ float part[256];
  int bl = blockIdx.x, t = threadIdx.x;
  for (int i = t; i < DINNER; i += 256) srow[i] = xi[(size_t)bl * DINNER + i];
  __syncthreads();
  int r = t & 63, seg = t >> 6;
  const float* wr = xproj_w + r * DINNER + seg * 256;
  const float* xs = srow + seg * 256;
  float s = 0.f;
  for (int k = 0; k < 256; ++k) s += xs[k] * wr[k];
  part[t] = s;
  __syncthreads();
  if (t < 64) dbc[bl * 64 + r] = part[r] + part[64 + r] + part[128 + r] + part[192 + r];
}

// ---------------------------------------------------------------------------
// Selective scan: thread = (b, channel d). 16 states in registers; sequential L.
// Writes gated output directly as bf16 (A operand of out_proj GEMM).
// ---------------------------------------------------------------------------
__global__ void mamba_scan_kernel(const float* __restrict__ dbc,
                                  const float* __restrict__ xi,
                                  const float* __restrict__ xz,
                                  const float* __restrict__ dt_w,
                                  const float* __restrict__ dt_b,
                                  const float* __restrict__ A_log,
                                  const float* __restrict__ Dp,
                                  unsigned short* __restrict__ ygate_bf)
{
  int b = blockIdx.x >> 2;
  int d = ((blockIdx.x & 3) << 8) + threadIdx.x;
  __shared__ float sdbc[64];

  float dtw[32];
  #pragma unroll
  for (int r = 0; r < 32; ++r) dtw[r] = dt_w[d * 32 + r];
  float Av[DSTATE];
  #pragma unroll
  for (int s = 0; s < DSTATE; ++s) Av[s] = -__expf(A_log[d * DSTATE + s]);
  float h[DSTATE];
  #pragma unroll
  for (int s = 0; s < DSTATE; ++s) h[s] = 0.f;
  const float dtb = dt_b[d];
  const float Dv  = Dp[d];

  for (int l = 0; l < SEQ; ++l) {
    int bl = b * SEQ + l;
    __syncthreads();
    if (threadIdx.x < 64) sdbc[threadIdx.x] = dbc[bl * 64 + threadIdx.x];
    __syncthreads();
    float dr = dtb;
    #pragma unroll
    for (int r = 0; r < 32; ++r) dr += sdbc[r] * dtw[r];
    float delta = (dr > 20.f) ? dr : log1pf(__expf(dr));   // softplus
    float xiv = xi[(size_t)bl * DINNER + d];
    float z   = xz[(size_t)bl * 2048 + DINNER + d];
    float dx  = delta * xiv;
    float y = 0.f;
    #pragma unroll
    for (int s = 0; s < DSTATE; ++s) {
      float dA = __expf(delta * Av[s]);
      h[s] = h[s] * dA + dx * sdbc[32 + s];
      y += h[s] * sdbc[48 + s];
    }
    y += Dv * xiv;
    float sz = z / (1.f + __expf(-z));
    ygate_bf[(size_t)bl * DINNER + d] = f2bf(y * sz);
  }
}

// ---------------------------------------------------------------------------
// Mean-pool tokens -> xpool[b, d]
// ---------------------------------------------------------------------------
__global__ void pool_kernel(const float* __restrict__ x, float* __restrict__ xp)
{
  int b = blockIdx.x, d = threadIdx.x;
  float s = 0.f;
  for (int l = 0; l < SEQ; ++l) s += x[(size_t)(b * SEQ + l) * DIM + d];
  xp[b * DIM + d] = s * (1.f / (float)SEQ);
}

// ---------------------------------------------------------------------------
// Build whole_bf[b, c(192), p(25600)] bf16 from mri/pet via LDS transpose.
// grid: (400 p-tiles, 8 = b*2+src)
// ---------------------------------------------------------------------------
__global__ void build_whole_kernel(const float* __restrict__ mri,
                                   const float* __restrict__ pet,
                                   unsigned short* __restrict__ whole)
{
  __shared__ unsigned short tile[96][65];
  int pt = blockIdx.x;
  int b = blockIdx.y >> 1, src = blockIdx.y & 1;
  const float* s = (src ? pet : mri) + (size_t)b * 2457600;
  int t = threadIdx.x, p0 = pt * 64;
  for (int i = t; i < 96 * 64; i += 256) {
    int ch = i % 96, pl = i / 96;                     // contiguous channel reads
    tile[ch][pl] = f2bf(s[(size_t)(p0 + pl) * 96 + ch]);
  }
  __syncthreads();
  for (int i = t; i < 96 * 64; i += 256) {
    int pl = i & 63, ch = i >> 6;                      // contiguous p writes
    whole[(size_t)((b * NKEY) + src * 96 + ch) * DCROSS + p0 + pl] = tile[ch][pl];
  }
}

// ---------------------------------------------------------------------------
// GEMV: one wave per output. out[b,n] = x[b,:]·W[n,:] (+bias)(+residual)
// ---------------------------------------------------------------------------
__global__ void gemv_kernel(const float* __restrict__ x,
                            const float* __restrict__ W,
                            const float* __restrict__ bias,
                            const float* __restrict__ res,
                            float* __restrict__ out,
                            int Brows, int N, int K)
{
  int idx = blockIdx.x * blockDim.x + threadIdx.x;
  int wave = idx >> 5, lane = idx & 31;
  if (wave >= Brows * N) return;
  int b = wave / N, n = wave - b * N;
  const float* xr = x + (size_t)b * K;
  const float* wr = W + (size_t)n * K;
  float p = 0.f;
  for (int k = lane; k < K; k += 32) p += xr[k] * wr[k];
  #pragma unroll
  for (int off = 16; off > 0; off >>= 1) p += __shfl_xor(p, off, 32);
  if (lane == 0) {
    float o = p;
    if (bias) o += bias[n];
    if (res)  o += res[(size_t)b * N + n];
    out[(size_t)b * N + n] = o;
  }
}

// ---------------------------------------------------------------------------
// Single-query attention per (b, head): logits over 192 keys, softmax, PV.
// ---------------------------------------------------------------------------
__global__ void attn_kernel(const float* __restrict__ q,
                            const float* __restrict__ k,
                            const float* __restrict__ v,
                            float* __restrict__ o)
{
  __shared__ float sl[NKEY];
  __shared__ float red;
  int b = blockIdx.x >> 3, h = blockIdx.x & 7, t = threadIdx.x;
  const float* qp = q + b * DIM + h * 64;
  const float* kp = k + (size_t)(b * NKEY + t) * DIM + h * 64;
  float lg = 0.f;
  #pragma unroll 8
  for (int d = 0; d < 64; ++d) lg += qp[d] * kp[d];
  sl[t] = lg * 0.125f;                       // 1/sqrt(64)
  __syncthreads();
  if (t == 0) { float m = sl[0]; for (int c = 1; c < NKEY; ++c) m = fmaxf(m, sl[c]); red = m; }
  __syncthreads();
  float e = __expf(sl[t] - red);
  __syncthreads();
  sl[t] = e;
  __syncthreads();
  if (t == 0) { float s = 0.f; for (int c = 0; c < NKEY; ++c) s += sl[c]; red = 1.f / s; }
  __syncthreads();
  if (t < 64) {
    float acc = 0.f;
    for (int c = 0; c < NKEY; ++c)
      acc += sl[c] * v[(size_t)(b * NKEY + c) * DIM + h * 64 + t];
    o[b * DIM + h * 64 + t] = acc * red;
  }
}

// ---------------------------------------------------------------------------
// LayerNorm rows of width 512
// ---------------------------------------------------------------------------
__global__ void layernorm_kernel(const float* __restrict__ x,
                                 const float* __restrict__ g,
                                 const float* __restrict__ bb,
                                 float* __restrict__ out)
{
  __shared__ float red[256];
  __shared__ float mu, inv;
  int b = blockIdx.x, t = threadIdx.x;
  float s = 0.f;
  for (int i = t; i < DIM; i += 256) s += x[b * DIM + i];
  red[t] = s; __syncthreads();
  for (int off = 128; off > 0; off >>= 1) { if (t < off) red[t] += red[t + off]; __syncthreads(); }
  if (t == 0) mu = red[0] / (float)DIM;
  __syncthreads();
  float m = mu; s = 0.f;
  for (int i = t; i < DIM; i += 256) { float d = x[b * DIM + i] - m; s += d * d; }
  red[t] = s; __syncthreads();
  for (int off = 128; off > 0; off >>= 1) { if (t < off) red[t] += red[t + off]; __syncthreads(); }
  if (t == 0) inv = rsqrtf(red[0] / (float)DIM + 1e-5f);
  __syncthreads();
  float iv = inv;
  for (int i = t; i < DIM; i += 256)
    out[b * DIM + i] = (x[b * DIM + i] - m) * iv * g[i] + bb[i];
}

// ---------------------------------------------------------------------------
// GEGLU gate: hg = a * gelu_exact(g)
// ---------------------------------------------------------------------------
__global__ void geglu_kernel(const float* __restrict__ h1, float* __restrict__ hg)
{
  int idx = blockIdx.x * blockDim.x + threadIdx.x;
  if (idx >= BSZ * 1024) return;
  int b = idx >> 10, k = idx & 1023;
  float a = h1[b * FFH + k];
  float g = h1[b * FFH + 1024 + k];
  hg[idx] = a * (0.5f * g * (1.f + erff(g * 0.70710678118654752f)));
}

// ---------------------------------------------------------------------------
// Host-side orchestration. Input indices assume JAX tree_flatten order
// (dicts sorted by key, DFS): feature_img, mri_cond, params{...}, pet_cond,
// x_categ, x_numer. Within params, keys sorted alphabetically; layers list
// expands in place with per-layer dicts also key-sorted.
// ---------------------------------------------------------------------------
extern "C" void kernel_launch(void* const* d_in, const int* in_sizes, int n_in,
                              void* d_out, int out_size, void* d_ws, size_t ws_size,
                              hipStream_t stream) {
  (void)in_sizes; (void)n_in; (void)out_size; (void)ws_size;

  const float* feature_img = (const float*)d_in[0];
  const float* mri         = (const float*)d_in[1];
  const float* categ_embed = (const float*)d_in[2];
  const float* cls         = (const float*)d_in[3];
  const float* ff_b1       = (const float*)d_in[4];
  const float* ff_b2       = (const float*)d_in[5];
  const float* ff_ln_b     = (const float*)d_in[6];
  const float* ff_ln_g     = (const float*)d_in[7];
  const float* ff_w1       = (const float*)d_in[8];
  const float* ff_w2       = (const float*)d_in[9];
  const float* k_b         = (const float*)d_in[10];
  const float* k_w         = (const float*)d_in[11];
  // layers at 12 + 10*i: A_log, D, conv_b, conv_w, dt_b, dt_w, in_w, norm_w, out_w, xproj_w
  const float* ln_b        = (const float*)d_in[72];
  const float* ln_g        = (const float*)d_in[73];
  const float* logit_b     = (const float*)d_in[74];
  const float* logit_w     = (const float*)d_in[75];
  const float* num_b       = (const float*)d_in[76];
  const float* num_w       = (const float*)d_in[77];
  const float* o_b         = (const float*)d_in[78];
  const float* o_w         = (const float*)d_in[79];
  const float* q_b         = (const float*)d_in[80];
  const float* q_w         = (const float*)d_in[81];
  const float* v_b         = (const float*)d_in[82];
  const float* v_w         = (const float*)d_in[83];
  const float* pet         = (const float*)d_in[84];
  const int*   x_categ     = (const int*)d_in[85];
  const float* x_numer     = (const float*)d_in[86];

  // workspace carve-up
  char* ws = (char*)d_ws;
  size_t off = 0;
  auto carve = [&](size_t bytes) { char* p = ws + off; off = (off + bytes + 255) & ~(size_t)255; return p; };
  float* x        = (float*)carve((size_t)ROWS * DIM * 4);
  float* xz       = (float*)carve((size_t)ROWS * 2048 * 4);
  float* xi       = (float*)carve((size_t)ROWS * DINNER * 4);
  float* dbc      = (float*)carve((size_t)ROWS * 64 * 4);
  float* kout     = (float*)carve((size_t)BSZ * NKEY * DIM * 4);
  float* vout     = (float*)carve((size_t)BSZ * NKEY * DIM * 4);
  float* part_k   = (float*)carve((size_t)KVSPLIT * BSZ * NKEY * DIM * 4);
  float* part_v   = (float*)carve((size_t)KVSPLIT * BSZ * NKEY * DIM * 4);
  float* xpool    = (float*)carve((size_t)BSZ * DIM * 4);
  float* qv       = (float*)carve((size_t)BSZ * DIM * 4);
  float* attn_o   = (float*)carve((size_t)BSZ * DIM * 4);
  float* x1       = (float*)carve((size_t)BSZ * DIM * 4);
  float* xln      = (float*)carve((size_t)BSZ * DIM * 4);
  float* h1       = (float*)carve((size_t)BSZ * FFH * 4);
  float* hg       = (float*)carve((size_t)BSZ * 1024 * 4);
  float* x2       = (float*)carve((size_t)BSZ * DIM * 4);
  float* ln2      = (float*)carve((size_t)BSZ * DIM * 4);
  unsigned short* xn_bf    = (unsigned short*)carve((size_t)ROWS * DIM * 2);
  unsigned short* ygate_bf = (unsigned short*)carve((size_t)ROWS * DINNER * 2);
  unsigned short* whole_bf = (unsigned short*)carve((size_t)BSZ * NKEY * DCROSS * 2);

  // 1. embed tokens
  embed_kernel<<<(ROWS * DIM + 255) / 256, 256, 0, stream>>>(
      x_categ, x_numer, feature_img, categ_embed, num_w, num_b, cls, x);

  // 2. build bf16 'whole' (B,192,25600)
  build_whole_kernel<<<dim3(DCROSS / 64, BSZ * 2), 256, 0, stream>>>(mri, pet, whole_bf);

  // 3. mamba layers
  for (int L = 0; L < 6; ++L) {
    const float* A_log  = (const float*)d_in[12 + 10 * L + 0];
    const float* Dp     = (const float*)d_in[12 + 10 * L + 1];
    const float* conv_b = (const float*)d_in[12 + 10 * L + 2];
    const float* conv_w = (const float*)d_in[12 + 10 * L + 3];
    const float* dt_b   = (const float*)d_in[12 + 10 * L + 4];
    const float* dt_w   = (const float*)d_in[12 + 10 * L + 5];
    const float* in_w   = (const float*)d_in[12 + 10 * L + 6];
    const float* norm_w = (const float*)d_in[12 + 10 * L + 7];
    const float* out_w  = (const float*)d_in[12 + 10 * L + 8];
    const float* xproj  = (const float*)d_in[12 + 10 * L + 9];

    rmsnorm_kernel<<<ROWS, 256, 0, stream>>>(x, norm_w, xn_bf);
    gemm_bf16_wmma<<<dim3(2048 / 64, (ROWS + 63) / 64, 1), 256, 0, stream>>>(
        xn_bf, DIM, in_w, DIM, nullptr, nullptr, xz, ROWS, 2048, DIM, 1);
    conv_silu_kernel<<<(ROWS * DINNER + 255) / 256, 256, 0, stream>>>(xz, conv_w, conv_b, xi);
    xproj_kernel<<<ROWS, 256, 0, stream>>>(xi, xproj, dbc);
    mamba_scan_kernel<<<BSZ * 4, 256, 0, stream>>>(dbc, xi, xz, dt_w, dt_b, A_log, Dp, ygate_bf);
    gemm_bf16_wmma<<<dim3(DIM / 64, (ROWS + 63) / 64, 1), 256, 0, stream>>>(
        ygate_bf, DINNER, out_w, DINNER, nullptr, /*residual=*/x, x, ROWS, DIM, DINNER, 1);
  }

  // 4. mean-pool tokens
  pool_kernel<<<BSZ, DIM, 0, stream>>>(x, xpool);

  // 5. q projection
  gemv_kernel<<<(BSZ * DIM * 32 + 255) / 256, 256, 0, stream>>>(
      xpool, q_w, q_b, nullptr, qv, BSZ, DIM, DIM);

  // 6. K/V projections: (768 x 25600) @ (512 x 25600)^T, split-K=4, bf16 WMMA
  gemm_bf16_wmma<<<dim3(DIM / 64, (BSZ * NKEY) / 64, KVSPLIT), 256, 0, stream>>>(
      whole_bf, DCROSS, k_w, DCROSS, nullptr, nullptr, part_k,
      BSZ * NKEY, DIM, DCROSS, KVSPLIT);
  gemm_bf16_wmma<<<dim3(DIM / 64, (BSZ * NKEY) / 64, KVSPLIT), 256, 0, stream>>>(
      whole_bf, DCROSS, v_w, DCROSS, nullptr, nullptr, part_v,
      BSZ * NKEY, DIM, DCROSS, KVSPLIT);
  splitk_reduce_kernel<<<(BSZ * NKEY * DIM + 255) / 256, 256, 0, stream>>>(
      part_k, k_b, nullptr, kout, BSZ * NKEY * DIM, DIM, KVSPLIT);
  splitk_reduce_kernel<<<(BSZ * NKEY * DIM + 255) / 256, 256, 0, stream>>>(
      part_v, v_b, nullptr, vout, BSZ * NKEY * DIM, DIM, KVSPLIT);

  // 7. attention (single query token per (b,h))
  attn_kernel<<<BSZ * 8, NKEY, 0, stream>>>(qv, kout, vout, attn_o);

  // 8. output projection + residual
  gemv_kernel<<<(BSZ * DIM * 32 + 255) / 256, 256, 0, stream>>>(
      attn_o, o_w, o_b, xpool, x1, BSZ, DIM, DIM);

  // 9. GEGLU feed-forward + residual
  layernorm_kernel<<<BSZ, 256, 0, stream>>>(x1, ff_ln_g, ff_ln_b, xln);
  gemv_kernel<<<(BSZ * FFH * 32 + 255) / 256, 256, 0, stream>>>(
      xln, ff_w1, ff_b1, nullptr, h1, BSZ, FFH, DIM);
  geglu_kernel<<<(BSZ * 1024 + 255) / 256, 256, 0, stream>>>(h1, hg);
  gemv_kernel<<<(BSZ * DIM * 32 + 255) / 256, 256, 0, stream>>>(
      hg, ff_w2, ff_b2, x1, x2, BSZ, DIM, 1024);

  // 10. final LN + logit
  layernorm_kernel<<<BSZ, 256, 0, stream>>>(x2, ln_g, ln_b, ln2);
  gemv_kernel<<<1, 256, 0, stream>>>(
      ln2, logit_w, logit_b, nullptr, (float*)d_out, BSZ, 1, DIM);
}
